// GatNetv1_73804718014638
// MI455X (gfx1250) — matrix operand order repaired and checked by reading, not compiled, not run on hardware
//
#include <hip/hip_runtime.h>
#include <hip/hip_bf16.h>

// ---------------------------------------------------------------------------
// GAT (2-layer, dense-mask) for MI455X / gfx1250, wave32.
// All GEMM-shaped math -> v_wmma_f32_16x16x32_bf16 (f32 accumulate).
// Operands pre-converted to bf16 ONCE (A row-major, B transposed) so WMMA
// fragments are built with 8 packed dword loads per lane, no in-loop cvt.
// Attention matrix never materialized: two-pass masked softmax with on-the-fly
// exp recompute inside the AV WMMA loop (saves ~0.5 GB of HBM round-trip).
// Adjacency is a 4096x4096 bitmask (2 MB, L2-resident).
// ---------------------------------------------------------------------------

#define N_NODES 4096
#define IN_DIM  512
#define HID     64
#define HEADS1  8
#define F_1     8
#define OUT_DIM 16
#define MWORDS  128          // 4096 / 32 mask words per row
#define NEG_BIG (-1.0e9f)
#define SLOPE   0.2f

typedef __attribute__((ext_vector_type(16))) __bf16 v16bf;
typedef __attribute__((ext_vector_type(8)))  float  v8f;

union FragBF { v16bf v; unsigned int u[8]; };

// ---- CDNA5 wave32 bf16 fragment index helpers (CDNA5 ISA) ------
// A (16x32, MxK): lane&15 = row M; lanes 0-15 carry K in {0..7,16..23},
// lanes 16-31 carry K in {8..15,24..31}. VGPR v holds halfword pair (K,K+1):
__device__ __forceinline__ int a_pair_base(int lane, int v) {
    return ((v < 4) ? (2 * v) : (16 + 2 * (v - 4))) + ((lane >= 16) ? 8 : 0);
}
// B (32x16, KxN): lane&15 = column N; K order matches A: lanes 0-15 carry
// K in {0..7,16..23}, lanes 16-31 carry K in {8..15,24..31}.
__device__ __forceinline__ int b_pair_base(int lane, int v) {
    return ((v < 4) ? (2 * v) : (16 + 2 * (v - 4))) + ((lane >= 16) ? 8 : 0);
}
// Per-halfword K for the scalar A-fragment build in the attention kernel.
__device__ __forceinline__ int a_frag_k(int lane, int i) {
    return a_pair_base(lane, i >> 1) + (i & 1);
}

// ---------------------------------------------------------------------------
// bf16 staging: row-major copy (A operands) and transposed copy (B operands).
// ---------------------------------------------------------------------------
__global__ void cvt_bf16(const float* __restrict__ src, __bf16* __restrict__ dst,
                         int n) {
    int t = blockIdx.x * blockDim.x + threadIdx.x;
    if (t < n) dst[t] = (__bf16)src[t];
}

// dst[c*rows + r] = src[r*cols + c]   (B operand -> column-major bf16)
__global__ void cvt_bf16_t(const float* __restrict__ src, __bf16* __restrict__ dst,
                           int rows, int cols) {
    int t = blockIdx.x * blockDim.x + threadIdx.x;
    if (t >= rows * cols) return;
    int r = t / cols, c = t % cols;
    dst[(size_t)c * rows + r] = (__bf16)src[t];
}

// ---------------------------------------------------------------------------
// Adjacency bitmask build
// ---------------------------------------------------------------------------
__global__ void zero_u32(unsigned int* __restrict__ p, int n) {
    int t = blockIdx.x * blockDim.x + threadIdx.x;
    if (t < n) p[t] = 0u;
}

__global__ void build_adj(const long long* __restrict__ ei,
                          unsigned int* __restrict__ mask, int E) {
    int e = blockIdx.x * blockDim.x + threadIdx.x;
    if (e >= E) return;
    int u = (int)ei[e];
    int v = (int)ei[E + e];
    atomicOr(&mask[(size_t)u * MWORDS + (v >> 5)], 1u << (v & 31));
    atomicOr(&mask[(size_t)v * MWORDS + (u >> 5)], 1u << (u & 31));
}

// ---------------------------------------------------------------------------
// C[M,Ncol] = A[M,K] x B[K,Ncol], A bf16 row-major, Bt bf16 [Ncol,K]
// (transposed). One wave per 16x16 output tile; fragments via packed dwords.
// ---------------------------------------------------------------------------
__global__ void gemm_bf16_wmma(const __bf16* __restrict__ A,
                               const __bf16* __restrict__ Bt,
                               float* __restrict__ C,
                               int M, int K, int Ncol) {
    int gwave = (blockIdx.x * blockDim.x + threadIdx.x) >> 5;
    int lane  = threadIdx.x & 31;
    int ntile = Ncol >> 4;
    if (gwave >= (M >> 4) * ntile) return;
    int tm = gwave / ntile;
    int tn = gwave % ntile;
    int m  = lane & 15;                      // A row within tile == B column

    const __bf16* arow = A  + (size_t)(tm * 16 + m) * K;
    const __bf16* bcol = Bt + (size_t)(tn * 16 + m) * K;

    v8f acc = {};
    for (int k0 = 0; k0 < K; k0 += 32) {
        FragBF a, b;
#pragma unroll
        for (int v = 0; v < 8; ++v) {
            a.u[v] = *(const unsigned int*)(arow + k0 + a_pair_base(lane, v));
            b.u[v] = *(const unsigned int*)(bcol + k0 + b_pair_base(lane, v));
        }
        acc = __builtin_amdgcn_wmma_f32_16x16x32_bf16(
            false, a.v, false, b.v, (short)0, acc, false, false);
    }
    int rbase = tm * 16 + ((lane >= 16) ? 8 : 0);
    int col   = tn * 16 + (lane & 15);
#pragma unroll
    for (int v = 0; v < 8; ++v)
        C[(size_t)(rbase + v) * Ncol + col] = acc[v];
}

// ---------------------------------------------------------------------------
// es[h,n] = <Hfeat[n, h*Fh : (h+1)*Fh], a_src[h]>, ed likewise. (f32 feats)
// ---------------------------------------------------------------------------
__global__ void attn_coeffs(const float* __restrict__ Hfeat,
                            const float* __restrict__ a_src,
                            const float* __restrict__ a_dst,
                            float* __restrict__ es, float* __restrict__ ed,
                            int heads, int Fh, int ldh) {
    int t = blockIdx.x * blockDim.x + threadIdx.x;
    if (t >= heads * N_NODES) return;
    int head = t / N_NODES, n = t % N_NODES;
    const float* hp = Hfeat + (size_t)n * ldh + head * Fh;
    const float* as = a_src + head * Fh;
    const float* ad = a_dst + head * Fh;
    float s = 0.f, d = 0.f;
    for (int f = 0; f < Fh; ++f) { float v = hp[f]; s += v * as[f]; d += v * ad[f]; }
    es[head * N_NODES + n] = s;
    ed[head * N_NODES + n] = d;
}

// ---------------------------------------------------------------------------
// Pass 1 of masked softmax: per (head,row) max and 1/sum(exp). One 128-thread
// block per (head,row); unmasked entries contribute NEG_BIG exactly like the
// reference's jnp.where (uniform fallback for isolated nodes preserved).
// ---------------------------------------------------------------------------
__global__ void softmax_stats(const unsigned int* __restrict__ mask,
                              const float* __restrict__ es,
                              const float* __restrict__ ed,
                              float* __restrict__ mrow,
                              float* __restrict__ rlrow) {
    int hr   = blockIdx.x;            // head * N + row   (N = 2^12)
    int row  = hr & (N_NODES - 1);
    int head = hr >> 12;
    float esv = es[hr];
    const unsigned int* mw  = mask + (size_t)row * MWORDS;
    const float*        edh = ed + head * N_NODES;

    __shared__ float red[128];
    float lmax = NEG_BIG;
    for (int j = threadIdx.x; j < N_NODES; j += 128) {
        float e = esv + edh[j];
        e = (e > 0.f) ? e : SLOPE * e;
        e = ((mw[j >> 5] >> (j & 31)) & 1u) ? e : NEG_BIG;
        lmax = fmaxf(lmax, e);
    }
    red[threadIdx.x] = lmax; __syncthreads();
    for (int s = 64; s > 0; s >>= 1) {
        if (threadIdx.x < s) red[threadIdx.x] = fmaxf(red[threadIdx.x], red[threadIdx.x + s]);
        __syncthreads();
    }
    float rmax = red[0]; __syncthreads();

    float lsum = 0.f;
    for (int j = threadIdx.x; j < N_NODES; j += 128) {
        float e = esv + edh[j];
        e = (e > 0.f) ? e : SLOPE * e;
        e = ((mw[j >> 5] >> (j & 31)) & 1u) ? e : NEG_BIG;
        lsum += __expf(e - rmax);
    }
    red[threadIdx.x] = lsum; __syncthreads();
    for (int s = 64; s > 0; s >>= 1) {
        if (threadIdx.x < s) red[threadIdx.x] += red[threadIdx.x + s];
        __syncthreads();
    }
    if (threadIdx.x == 0) { mrow[hr] = rmax; rlrow[hr] = 1.0f / red[0]; }
}

// ---------------------------------------------------------------------------
// Pass 2: Out[i, head*Fh+f] = sum_j softmax(e)_ij * Hfeat[j, head*Fh+f]
// One wave per (head, 16-row tile). Probabilities recomputed on the fly
// (v_exp_f32 -> v_cvt_pk_bf16_f32) -> A fragment; B fragment from the
// TRANSPOSED bf16 feature matrix Ht [heads*Fh, N] with packed dword loads
// (Fh columns zero-padded to 16). f32 WMMA accumulate; optional fused ELU.
// ---------------------------------------------------------------------------
__global__ void attn_av_wmma(const unsigned int* __restrict__ mask,
                             const float* __restrict__ es,
                             const float* __restrict__ ed,
                             const float* __restrict__ mrow,
                             const float* __restrict__ rlrow,
                             const __bf16* __restrict__ Ht,   // [heads*Fh, N]
                             float* __restrict__ Out,         // [N, ldo]
                             int heads, int Fh, int ldo,
                             int applyElu) {
    int gwave = (blockIdx.x * blockDim.x + threadIdx.x) >> 5;
    int lane  = threadIdx.x & 31;
    const int tilesPerHead = N_NODES >> 4;          // 256
    if (gwave >= heads * tilesPerHead) return;
    int head = gwave / tilesPerHead;
    int tile = gwave % tilesPerHead;

    int m   = lane & 15;
    int row = tile * 16 + m;                        // this lane's A-row / B-col
    int hr  = head * N_NODES + row;
    float esv = es[hr], mv = mrow[hr], rl = rlrow[hr];
    const float*        edh = ed + head * N_NODES;
    const unsigned int* mw  = mask + (size_t)row * MWORDS;
    // B column pointer: column head*Fh+m of the feature matrix, contiguous in j
    const __bf16* bcol = Ht + (size_t)(head * Fh + (m < Fh ? m : 0)) * N_NODES;
    const bool colActive = (m < Fh);

    v8f acc = {};
    for (int j0 = 0; j0 < N_NODES; j0 += 32) {
        unsigned int w = mw[j0 >> 5];               // 32 mask bits == K-chunk
        FragBF a, b;
#pragma unroll
        for (int i = 0; i < 16; ++i) {
            int ka = a_frag_k(lane, i);
            float e = esv + edh[j0 + ka];
            e = (e > 0.f) ? e : SLOPE * e;
            e = ((w >> ka) & 1u) ? e : NEG_BIG;
            a.v[i] = (__bf16)(__expf(e - mv) * rl);
        }
#pragma unroll
        for (int v = 0; v < 8; ++v)
            b.u[v] = colActive
                ? *(const unsigned int*)(bcol + j0 + b_pair_base(lane, v)) : 0u;
        if (j0 + 32 < N_NODES)                      // global_prefetch_b8
            __builtin_prefetch(bcol + j0 + 32, 0, 1);
        acc = __builtin_amdgcn_wmma_f32_16x16x32_bf16(
            false, a.v, false, b.v, (short)0, acc, false, false);
    }
    int rbase = tile * 16 + ((lane >= 16) ? 8 : 0);
    int col   = lane & 15;
    if (col < Fh) {
#pragma unroll
        for (int v = 0; v < 8; ++v) {
            float val = acc[v];
            if (applyElu) val = (val > 0.f) ? val : (__expf(val) - 1.0f);
            Out[(size_t)(rbase + v) * ldo + head * Fh + col] = val;
        }
    }
}

// ---------------------------------------------------------------------------
// Fused log_softmax over the 16 output logits per node.
// ---------------------------------------------------------------------------
__global__ void logsoftmax16(const float* __restrict__ X, float* __restrict__ Out) {
    int n = blockIdx.x * blockDim.x + threadIdx.x;
    if (n >= N_NODES) return;
    const float* x = X + (size_t)n * OUT_DIM;
    float mx = x[0];
#pragma unroll
    for (int c = 1; c < OUT_DIM; ++c) mx = fmaxf(mx, x[c]);
    float s = 0.f;
#pragma unroll
    for (int c = 0; c < OUT_DIM; ++c) s += __expf(x[c] - mx);
    float lse = mx + __logf(s);
#pragma unroll
    for (int c = 0; c < OUT_DIM; ++c) Out[(size_t)n * OUT_DIM + c] = x[c] - lse;
}

// ---------------------------------------------------------------------------
extern "C" void kernel_launch(void* const* d_in, const int* in_sizes, int n_in,
                              void* d_out, int out_size, void* d_ws, size_t ws_size,
                              hipStream_t stream) {
    (void)n_in; (void)out_size; (void)ws_size;

    const float*     x      = (const float*)d_in[0];
    const long long* eidx   = (const long long*)d_in[1];   // jnp.int64
    const float*     W1     = (const float*)d_in[2];
    const float*     a1_src = (const float*)d_in[3];
    const float*     a1_dst = (const float*)d_in[4];
    const float*     W2     = (const float*)d_in[5];
    const float*     a2_src = (const float*)d_in[6];
    const float*     a2_dst = (const float*)d_in[7];
    float*           out    = (float*)d_out;
    const int E = in_sizes[1] / 2;

    // ---- workspace layout (all offsets 4-byte aligned) ----
    char* ws = (char*)d_ws;
    const size_t maskW = (size_t)N_NODES * MWORDS;                  // u32 words
    unsigned int* mask = (unsigned int*)ws; ws += maskW * 4;        // 2 MB
    float* h1    = (float*)ws;  ws += (size_t)N_NODES * HID * 4;    // 1 MB
    float* es1   = (float*)ws;  ws += (size_t)HEADS1 * N_NODES * 4;
    float* ed1   = (float*)ws;  ws += (size_t)HEADS1 * N_NODES * 4;
    float* m1    = (float*)ws;  ws += (size_t)HEADS1 * N_NODES * 4;
    float* rl1   = (float*)ws;  ws += (size_t)HEADS1 * N_NODES * 4;
    float* helu  = (float*)ws;  ws += (size_t)N_NODES * HID * 4;    // 1 MB
    float* h2    = (float*)ws;  ws += (size_t)N_NODES * OUT_DIM * 4;
    float* es2   = (float*)ws;  ws += (size_t)N_NODES * 4;
    float* ed2   = (float*)ws;  ws += (size_t)N_NODES * 4;
    float* m2    = (float*)ws;  ws += (size_t)N_NODES * 4;
    float* rl2   = (float*)ws;  ws += (size_t)N_NODES * 4;
    float* out2  = (float*)ws;  ws += (size_t)N_NODES * OUT_DIM * 4;
    // bf16 staging buffers
    __bf16* xb    = (__bf16*)ws; ws += (size_t)N_NODES * IN_DIM * 2;  // 4 MB
    __bf16* w1t   = (__bf16*)ws; ws += (size_t)HID * IN_DIM * 2;      // [64,512]
    __bf16* h1t   = (__bf16*)ws; ws += (size_t)HID * N_NODES * 2;     // [64,4096]
    __bf16* helub = (__bf16*)ws; ws += (size_t)N_NODES * HID * 2;
    __bf16* w2t   = (__bf16*)ws; ws += (size_t)OUT_DIM * HID * 2;     // [16,64]
    __bf16* h2t   = (__bf16*)ws; ws += (size_t)OUT_DIM * N_NODES * 2; // [16,4096]

    // 0) adjacency bitmask
    zero_u32<<<(int)((maskW + 255) / 256), 256, 0, stream>>>(mask, (int)maskW);
    build_adj<<<(E + 255) / 256, 256, 0, stream>>>(eidx, mask, E);

    // 1) stage x (row-major bf16) and W1 (transposed bf16), once
    cvt_bf16<<<(N_NODES * IN_DIM) / 256, 256, 0, stream>>>(x, xb, N_NODES * IN_DIM);
    cvt_bf16_t<<<(IN_DIM * HID) / 256, 256, 0, stream>>>(W1, w1t, IN_DIM, HID);

    // 2) h1 = x @ W1   [4096,512] x [512,64]  -> 1024 waves
    gemm_bf16_wmma<<<(1024 * 32) / 256, 256, 0, stream>>>(xb, w1t, h1,
                                                          N_NODES, IN_DIM, HID);
    // 3) attention coefficients + transposed bf16 features for the AV pass
    attn_coeffs<<<(HEADS1 * N_NODES) / 256, 256, 0, stream>>>(
        h1, a1_src, a1_dst, es1, ed1, HEADS1, F_1, HID);
    cvt_bf16_t<<<(N_NODES * HID) / 256, 256, 0, stream>>>(h1, h1t, N_NODES, HID);

    // 4) masked softmax stats, layer 1 (one block per head-row)
    softmax_stats<<<HEADS1 * N_NODES, 128, 0, stream>>>(mask, es1, ed1, m1, rl1);
    // 5) AV via WMMA + fused ELU -> helu   (8 heads x 256 tiles = 2048 waves)
    attn_av_wmma<<<(2048 * 32) / 256, 256, 0, stream>>>(
        mask, es1, ed1, m1, rl1, h1t, helu, HEADS1, F_1, HID, 1);

    // 6) layer 2 feature transform: h2 = helu @ W2  [4096,64] x [64,16]
    cvt_bf16<<<(N_NODES * HID) / 256, 256, 0, stream>>>(helu, helub, N_NODES * HID);
    cvt_bf16_t<<<(HID * OUT_DIM) / 256, 256, 0, stream>>>(W2, w2t, HID, OUT_DIM);
    gemm_bf16_wmma<<<(256 * 32) / 256, 256, 0, stream>>>(helub, w2t, h2,
                                                         N_NODES, HID, OUT_DIM);
    // 7) attention coefficients + transposed features, layer 2
    attn_coeffs<<<N_NODES / 256, 256, 0, stream>>>(
        h2, a2_src, a2_dst, es2, ed2, 1, OUT_DIM, OUT_DIM);
    cvt_bf16_t<<<(N_NODES * OUT_DIM) / 256, 256, 0, stream>>>(h2, h2t,
                                                              N_NODES, OUT_DIM);
    // 8) masked softmax stats, layer 2
    softmax_stats<<<N_NODES, 128, 0, stream>>>(mask, es2, ed2, m2, rl2);
    // 9) AV via WMMA (1 head, F=16, mean over 1 head == identity)
    attn_av_wmma<<<(256 * 32) / 256, 256, 0, stream>>>(
        mask, es2, ed2, m2, rl2, h2t, out2, 1, OUT_DIM, OUT_DIM, 0);

    // 10) log_softmax -> d_out
    logsoftmax16<<<N_NODES / 256, 256, 0, stream>>>(out2, out);
}
